// VectorQuantizer_34411277976262
// MI455X (gfx1250) — matrix operand order repaired and checked by reading, not compiled, not run on hardware
//
#include <hip/hip_runtime.h>
#include <hip/hip_bf16.h>

// ---------------------------------------------------------------------------
// VQ forward for MI455X (gfx1250, wave32, WMMA bf16 path)
//   argmin_k ||z_n - e_k||^2  ==  argmin_k (||e_k||^2 - 2 z.e_k)
//   cross-term via v_wmma_f32_16x16x32_bf16; B staged through LDS with
//   double-buffered global_load_async_to_lds_b128 (ASYNCcnt), and the
//   LDS->VGPR frag stream software-pipelined 2 steps ahead.
// ---------------------------------------------------------------------------

#define N_ROWS   32768
#define K_CODES  8192
#define D_DIM    512
#define STEP     32          // codes staged per buffer step (2 WMMA column tiles)
#define NSTEP    (K_CODES / STEP)
#define BSTRIDE  520         // padded bf16 elements per code row in LDS (260 dwords)

typedef __attribute__((ext_vector_type(16))) __bf16 v16bf;
typedef __attribute__((ext_vector_type(8)))  float  v8f;
typedef __attribute__((ext_vector_type(4)))  unsigned int uint4v;

union Frag16 {
    uint4v u[2];
    v16bf  v;
};

__device__ __forceinline__ unsigned short f32_to_bf16_rne(float x) {
    unsigned int u = __float_as_uint(x);
    unsigned int lsb = (u >> 16) & 1u;
    u += 0x7FFFu + lsb;             // round-to-nearest-even
    return (unsigned short)(u >> 16);
}

// ---------------------------------------------------------------------------
// Pass A: codebook -> bf16, plus exact fp32 row norms ||e_k||^2
// grid: K_CODES blocks x 128 threads
// ---------------------------------------------------------------------------
__global__ __launch_bounds__(128) void vq_prep_emb(const float* __restrict__ emb,
                                                   unsigned short* __restrict__ emb_bf,
                                                   float* __restrict__ en2) {
    __shared__ float red[128];
    const int row = blockIdx.x;
    const int t   = threadIdx.x;
    const float* src = emb + (size_t)row * D_DIM;
    unsigned short* dst = emb_bf + (size_t)row * D_DIM;
    float s = 0.0f;
#pragma unroll
    for (int i = 0; i < 4; ++i) {
        int e = t + i * 128;
        float x = src[e];
        s += x * x;
        dst[e] = f32_to_bf16_rne(x);
    }
    red[t] = s;
    __syncthreads();
    for (int d = 64; d > 0; d >>= 1) {
        if (t < d) red[t] += red[t + d];
        __syncthreads();
    }
    if (t == 0) en2[row] = red[0];
}

// ---------------------------------------------------------------------------
// Pass B: z -> bf16 (elementwise)
// ---------------------------------------------------------------------------
__global__ __launch_bounds__(256) void vq_prep_z(const float* __restrict__ z,
                                                 unsigned short* __restrict__ z_bf) {
    const size_t total4 = (size_t)N_ROWS * D_DIM / 4;
    for (size_t i = (size_t)blockIdx.x * blockDim.x + threadIdx.x; i < total4;
         i += (size_t)gridDim.x * blockDim.x) {
        const float4 v = ((const float4*)z)[i];
        ushort4 o;
        o.x = f32_to_bf16_rne(v.x);
        o.y = f32_to_bf16_rne(v.y);
        o.z = f32_to_bf16_rne(v.z);
        o.w = f32_to_bf16_rne(v.w);
        ((ushort4*)z_bf)[i] = o;
    }
}

// ---------------------------------------------------------------------------
// Pass C: WMMA argmin sweep.
//  - one wave owns 16 z-rows; A tile cached in 128 VGPRs
//  - block shares 32-code B tiles staged via async global->LDS, double-buffered
//  - 32 WMMAs per barrier; B frags from LDS software-pipelined depth 2
// grid: 256 blocks x 256 threads (8 waves/block, 16 rows/wave)
// ---------------------------------------------------------------------------
__global__ __launch_bounds__(256) void vq_argmin(const unsigned short* __restrict__ z_bf,
                                                 const unsigned short* __restrict__ emb_bf,
                                                 const float* __restrict__ en2,
                                                 int* __restrict__ out_idx) {
    __shared__ unsigned short Bs[2][STEP * BSTRIDE];   // 2 x 32.5 KB

    const int tid  = threadIdx.x;
    const int lane = tid & 31;
    const int wave = tid >> 5;
    const int h    = lane >> 4;    // 16-lane half
    const int j    = lane & 15;    // row (A) / column (B, C)
    const int r0   = (blockIdx.x * 8 + wave) * 16;

    // Stage one 32-code step (32 KB) into LDS buffer `buf` asynchronously:
    // 2048 B128 transfers, 8 per thread. ASYNCcnt += 8 per thread.
    auto stage = [&](int c0, int buf) {
#pragma unroll
        for (int i = 0; i < 8; ++i) {
            const int m    = tid + i * 256;      // 0..2047
            const int code = m >> 6;             // 64 transfers per code row
            const int el   = (m & 63) << 3;      // bf16 element offset (x8)
            const unsigned long long ga =
                (unsigned long long)(uintptr_t)(emb_bf + (size_t)(c0 + code) * D_DIM + el);
            const unsigned lds =
                (unsigned)(uintptr_t)&Bs[buf][code * BSTRIDE + el];
            asm volatile("global_load_async_to_lds_b128 %0, %1, off"
                         :: "v"(lds), "v"(ga) : "memory");
        }
    };

    // Preload A: 16 k-steps of 32; lane half h holds k = 16h..16h+15 of each step.
    v16bf A[16];
    {
        const unsigned short* ap = z_bf + (size_t)(r0 + j) * D_DIM;
#pragma unroll
        for (int t = 0; t < 16; ++t) {
            const int kb = t * 32;
            Frag16 f;
            f.u[0] = *(const uint4v*)(ap + kb + h * 16);
            f.u[1] = *(const uint4v*)(ap + kb + h * 16 + 8);
            A[t] = f.v;
        }
    }

    // Prologue: stage step 0 and wait for it.
    stage(0, 0);
    asm volatile("s_wait_asynccnt 0x0" ::: "memory");
    __syncthreads();

    float minv[8];
    int   mini[8];
#pragma unroll
    for (int v = 0; v < 8; ++v) { minv[v] = 3.4e38f; mini[v] = 0; }

    for (int c = 0; c < NSTEP; ++c) {
        const int buf = c & 1;
        if (c + 1 < NSTEP) stage((c + 1) * STEP, buf ^ 1);   // overlap with compute

#pragma unroll
        for (int sub = 0; sub < 2; ++sub) {
            const int code = c * STEP + sub * 16 + j;        // this lane's column
            // B layout: lane j = column, half h holds K in {h*16 .. h*16+15}.
            const unsigned short* bp = &Bs[buf][(sub * 16 + j) * BSTRIDE + h * 16];

            // Software-pipelined frag stream, depth 2.
            Frag16 f[3];
            f[0].u[0] = *(const uint4v*)(bp);
            f[0].u[1] = *(const uint4v*)(bp + 8);
            f[1].u[0] = *(const uint4v*)(bp + 32);
            f[1].u[1] = *(const uint4v*)(bp + 40);

            v8f acc = {};
#pragma unroll
            for (int t = 0; t < 16; ++t) {
                if (t + 2 < 16) {
                    f[(t + 2) % 3].u[0] = *(const uint4v*)(bp + (t + 2) * 32);
                    f[(t + 2) % 3].u[1] = *(const uint4v*)(bp + (t + 2) * 32 + 8);
                }
                acc = __builtin_amdgcn_wmma_f32_16x16x32_bf16(
                    false, A[t], false, f[t % 3].v, (short)0, acc, false, false);
            }
            const float en = en2[code];
#pragma unroll
            for (int v = 0; v < 8; ++v) {
                const float s = en - 2.0f * acc[v];   // row = v + 8h, col = code
                if (s < minv[v]) { minv[v] = s; mini[v] = code; }
            }
        }

        // Next tile must be fully in LDS (and this tile fully read by all
        // waves) before the next step flips buffers.
        asm volatile("s_wait_asynccnt 0x0" ::: "memory");
        __syncthreads();
    }

    // Butterfly min+argmin across the 16 lanes of each half (xor stays in-half).
#pragma unroll
    for (int d = 1; d < 16; d <<= 1) {
#pragma unroll
        for (int v = 0; v < 8; ++v) {
            const float ov = __shfl_xor(minv[v], d, 32);
            const int   oi = __shfl_xor(mini[v], d, 32);
            if (ov < minv[v] || (ov == minv[v] && oi < mini[v])) {
                minv[v] = ov; mini[v] = oi;
            }
        }
    }
    if (j == 0) {
#pragma unroll
        for (int v = 0; v < 8; ++v)
            out_idx[r0 + h * 8 + v] = mini[v];
    }
}

// ---------------------------------------------------------------------------
// Pass D: gather z_q (fp32), straight-through output, deterministic partial
// sums of (z_q - z)^2 per block. grid: GL_BLOCKS x 256
// ---------------------------------------------------------------------------
#define GL_BLOCKS 256

__global__ __launch_bounds__(256) void vq_gather_loss(const float* __restrict__ z,
                                                      const float* __restrict__ emb,
                                                      const int* __restrict__ idx,
                                                      float* __restrict__ out_zq,
                                                      float* __restrict__ out_idxf,
                                                      float* __restrict__ partials) {
    __shared__ float red[256];
    const int t = threadIdx.x;
    float local = 0.0f;
    for (int row = blockIdx.x; row < N_ROWS; row += gridDim.x) {
        const int k = idx[row];
        const float* er = emb + (size_t)k * D_DIM;
        const float* zr = z + (size_t)row * D_DIM;
        float* orow = out_zq + (size_t)row * D_DIM;
#pragma unroll
        for (int i = 0; i < 2; ++i) {
            const int e = t + i * 256;
            const float zq = er[e];
            const float zv = zr[e];
            const float diff = zq - zv;     // matches reference rounding
            local += diff * diff;
            orow[e] = zv + diff;            // z + sg(z_q - z)
        }
        if (t == 0) out_idxf[row] = (float)k;
    }
    red[t] = local;
    __syncthreads();
    for (int d = 128; d > 0; d >>= 1) {
        if (t < d) red[t] += red[t + d];
        __syncthreads();
    }
    if (t == 0) partials[blockIdx.x] = red[0];
}

__global__ __launch_bounds__(256) void vq_final_loss(const float* __restrict__ partials,
                                                     float* __restrict__ out_loss) {
    __shared__ float red[256];
    const int t = threadIdx.x;
    red[t] = (t < GL_BLOCKS) ? partials[t] : 0.0f;
    __syncthreads();
    for (int d = 128; d > 0; d >>= 1) {
        if (t < d) red[t] += red[t + d];
        __syncthreads();
    }
    if (t == 0) {
        // loss = mean + BETA*mean = 2 * sum / (N*D)
        *out_loss = 2.0f * red[0] / (float)((size_t)N_ROWS * D_DIM);
    }
}

// ---------------------------------------------------------------------------
extern "C" void kernel_launch(void* const* d_in, const int* in_sizes, int n_in,
                              void* d_out, int out_size, void* d_ws, size_t ws_size,
                              hipStream_t stream) {
    const float* z   = (const float*)d_in[0];   // [32768, 512] fp32
    const float* emb = (const float*)d_in[1];   // [8192, 512]  fp32

    // Workspace layout (16B aligned)
    char* ws = (char*)d_ws;
    unsigned short* emb_bf = (unsigned short*)(ws);                         // 8 MB
    unsigned short* z_bf   = (unsigned short*)(ws + 8388608);               // 32 MB
    float*          en2    = (float*)(ws + 8388608 + 33554432);             // 32 KB
    int*            idx_ws = (int*)(ws + 8388608 + 33554432 + 32768);       // 128 KB
    float*          parts  = (float*)(ws + 8388608 + 33554432 + 32768 + 131072);

    float* out      = (float*)d_out;
    float* out_loss = out;                                  // [1]
    float* out_zq   = out + 1;                              // [32768*512]
    float* out_idxf = out + 1 + (size_t)N_ROWS * D_DIM;     // [32768]

    vq_prep_emb<<<K_CODES, 128, 0, stream>>>(emb, emb_bf, en2);
    vq_prep_z<<<4096, 256, 0, stream>>>(z, z_bf);
    vq_argmin<<<N_ROWS / 128, 256, 0, stream>>>(z_bf, emb_bf, en2, idx_ws);
    vq_gather_loss<<<GL_BLOCKS, 256, 0, stream>>>(z, emb, idx_ws, out_zq, out_idxf, parts);
    vq_final_loss<<<1, 256, 0, stream>>>(parts, out_loss);
}
